// ConvAttention_19327352832680
// MI455X (gfx1250) — matrix operand order repaired and checked by reading, not compiled, not run on hardware
//
#include <hip/hip_runtime.h>
#include <hip/hip_fp16.h>

typedef __attribute__((ext_vector_type(16))) _Float16 v16h;
typedef __attribute__((ext_vector_type(8)))  float    v8f;

#define DIMC  256
#define IMGSZ 56
#define NTOK  3136
#define NHEAD 8
#define DHEAD 64
#define INNER 512
#define BATCH 4
#define TROWS (BATCH * NTOK)   // 12544
#define KBLKS (NTOK / 64)      // 49

// ---------------------------------------------------------------------------
// Stage 1: depthwise 3x3 conv + BN(eval) + exact GELU -> f16 activations
// grid: B*NTOK blocks, 256 threads (one per channel); coalesced over channels.
// ---------------------------------------------------------------------------
__global__ __launch_bounds__(256) void dwconv_bn_gelu(
        const float* __restrict__ x,
        const float* __restrict__ dww,
        const float* __restrict__ dwb,
        const float* __restrict__ bng,
        const float* __restrict__ bnb,
        const float* __restrict__ bnm,
        const float* __restrict__ bnv,
        __half* __restrict__ y)
{
    const int c   = threadIdx.x;
    const int blk = blockIdx.x;            // b*NTOK + p
    const int b   = blk / NTOK;
    const int p   = blk - b * NTOK;
    const int hh  = p / IMGSZ;
    const int ww  = p - hh * IMGSZ;
    const float* xb = x + (size_t)b * NTOK * DIMC;

    float acc = 0.f;
#pragma unroll
    for (int ky = 0; ky < 3; ky++) {
        const int yy = hh + ky - 1;
        if (yy < 0 || yy >= IMGSZ) continue;
#pragma unroll
        for (int kx = 0; kx < 3; kx++) {
            const int xx = ww + kx - 1;
            if (xx < 0 || xx >= IMGSZ) continue;
            acc += dww[c * 9 + ky * 3 + kx] * xb[(size_t)(yy * IMGSZ + xx) * DIMC + c];
        }
    }
    acc += dwb[c];
    const float inv = bng[c] * rsqrtf(bnv[c] + 1e-5f);
    acc = acc * inv + (bnb[c] - bnm[c] * inv);
    acc = 0.5f * acc * (1.f + erff(acc * 0.70710678118654752f)); // exact GELU
    y[(size_t)blk * DIMC + c] = (__half)acc;
}

// ---------------------------------------------------------------------------
// WMMA GEMM:  C[t,o] = sum_k A[t,k] * W[o,k]  (+ bias[o])
//   A: f16 row-major [T, KDIM], W: f32 row-major [OCOLS, KDIM]
//   MODE 0: out f16 into [B, H, N, DH] head layout, scaled by alpha (q/k/v)
//   MODE 1: out f32 into [T, OCOLS] (final projection -> d_out)
// ---------------------------------------------------------------------------
template<int KDIM, int OCOLS, int MODE>
__global__ __launch_bounds__(128) void gemm_wmma(
        const __half* __restrict__ A,
        const float* __restrict__ W,
        const float* __restrict__ bias,
        void* __restrict__ outp,
        float alpha)
{
    __shared__ __align__(16) _Float16 As[64 * 40];  // 32-k chunk, padded stride
    __shared__ __align__(16) _Float16 Ws[64 * 40];

    const int tid  = threadIdx.x;        // 0..127
    const int wave = tid >> 5;
    const int lane = tid & 31;
    const int l16  = lane & 15;
    const int hi   = lane >> 4;          // 0 or 1
    const int rowblk = blockIdx.x * 64;
    const int oblk   = blockIdx.y * 64;

    v8f cacc[4] = {};

    for (int kc = 0; kc < KDIM / 32; kc++) {
        __syncthreads();
        // stage A tile [64 rows x 32 k] (already f16)
        for (int i = tid; i < 256; i += 128) {
            const int r = i >> 2, c4 = i & 3;
            const uint4 v = *(const uint4*)(A + (size_t)(rowblk + r) * KDIM + kc * 32 + c4 * 8);
            *(uint4*)(&As[r * 40 + c4 * 8]) = v;
        }
        // stage W tile [64 ocols x 32 k], f32 -> f16
        for (int i = tid; i < 2048; i += 128) {
            const int r = i >> 5, cc = i & 31;
            Ws[r * 40 + cc] = (_Float16)W[(size_t)(oblk + r) * KDIM + kc * 32 + cc];
        }
        __syncthreads();

        union Frag { uint4 u[2]; v16h v; };
        Frag af;   // A frag: lane row = wave*16+l16; halves hi*8..+7, +16..+23
        {
            const int ar = wave * 16 + l16;
            af.u[0] = *(const uint4*)(&As[ar * 40 + hi * 8]);
            af.u[1] = *(const uint4*)(&As[ar * 40 + hi * 8 + 16]);
        }
#pragma unroll
        for (int t = 0; t < 4; t++) {
            Frag bf;   // B frag: lane col = t*16+l16; halves hi*8..+7, +16..+23
            const int br = t * 16 + l16;
            bf.u[0] = *(const uint4*)(&Ws[br * 40 + hi * 8]);
            bf.u[1] = *(const uint4*)(&Ws[br * 40 + hi * 8 + 16]);
            cacc[t] = __builtin_amdgcn_wmma_f32_16x16x32_f16(
                false, af.v, false, bf.v, (short)0, cacc[t], false, false);
        }
    }

    // epilogue: C layout row = (lane<16?0:8)+j, col = l16
#pragma unroll
    for (int t = 0; t < 4; t++) {
#pragma unroll
        for (int j = 0; j < 8; j++) {
            const int row = rowblk + wave * 16 + hi * 8 + j;
            const int o   = oblk + t * 16 + l16;
            const float val = cacc[t][j] + bias[o];
            if (MODE == 0) {
                const int b = row / NTOK, n = row - b * NTOK;
                const int h = o >> 6,     d = o & 63;
                ((__half*)outp)[(((size_t)b * NHEAD + h) * NTOK + n) * DHEAD + d] =
                    (__half)(val * alpha);
            } else {
                ((float*)outp)[(size_t)row * OCOLS + o] = val;
            }
        }
    }
}

// ---------------------------------------------------------------------------
// Flash attention: one block per (b,h, 64-query tile). 128 threads = 4 waves,
// 16 query rows per wave; 49 key blocks of 64. Softmax denominator computed
// with an all-ones 5th V tile (P * 1 = row sums, replicated across lanes),
// so no sum shuffles and no separate l bookkeeping. SCALE pre-folded into q.
// ---------------------------------------------------------------------------
__global__ __launch_bounds__(128) void flash_attn(
        const __half* __restrict__ Q,
        const __half* __restrict__ K,
        const __half* __restrict__ V,
        __half* __restrict__ O)
{
    __shared__ __align__(16) _Float16 Qs [64 * 72];
    __shared__ __align__(16) _Float16 Ks [64 * 72];
    __shared__ __align__(16) _Float16 Vts[80 * 72];  // [dh][key]; rows 64..79 = 1.0
    __shared__ __align__(16) _Float16 Ps [64 * 72];  // 4 waves x 16 rows

    const int tid  = threadIdx.x;
    const int wave = tid >> 5;
    const int lane = tid & 31;
    const int l16  = lane & 15;
    const int hi   = lane >> 4;
    const int blk  = blockIdx.x;
    const int qb   = blk % KBLKS;
    const int bh   = blk / KBLKS;         // b*8 + h
    const size_t headoff = (size_t)bh * NTOK * DHEAD;

    // ones tile for the row-sum accumulator (written once, never overwritten)
    for (int i = tid; i < 16 * 72; i += 128)
        Vts[64 * 72 + i] = (_Float16)1.0f;

    // stage Q tile [64 rows x 64 dh]
    const __half* Qg = Q + headoff + (size_t)qb * 64 * DHEAD;
    for (int i = tid; i < 512; i += 128) {
        const int r = i >> 3, c8 = i & 7;
        const uint4 v = *(const uint4*)(Qg + r * DHEAD + c8 * 8);
        *(uint4*)(&Qs[r * 72 + c8 * 8]) = v;
    }
    __syncthreads();

    union Frag { uint4 u[2]; v16h v; };
    Frag qa[2];
#pragma unroll
    for (int c = 0; c < 2; c++) {
        const int ar = wave * 16 + l16;
        const int base = c * 32 + hi * 8;
        qa[c].u[0] = *(const uint4*)(&Qs[ar * 72 + base]);
        qa[c].u[1] = *(const uint4*)(&Qs[ar * 72 + base + 16]);
    }

    float mi[8];
    v8f acc[5] = {};          // acc[0..3] = O tiles, acc[4] = row-sum tile
#pragma unroll
    for (int j = 0; j < 8; j++) mi[j] = -1e30f;

    for (int kb = 0; kb < KBLKS; kb++) {
        __syncthreads();   // previous iteration done reading Ks/Vts
        const __half* Kg = K + headoff + (size_t)kb * 64 * DHEAD;
        const __half* Vg = V + headoff + (size_t)kb * 64 * DHEAD;
        // prefetch next key block while this one computes (64B per lane = 8KB)
        if (kb + 1 < KBLKS) {
            __builtin_prefetch(Kg + 64 * DHEAD + tid * 32, 0, 0);
            __builtin_prefetch(Vg + 64 * DHEAD + tid * 32, 0, 0);
        }
        for (int i = tid; i < 512; i += 128) {
            const int r = i >> 3, c8 = i & 7;
            const uint4 kv = *(const uint4*)(Kg + r * DHEAD + c8 * 8);
            *(uint4*)(&Ks[r * 72 + c8 * 8]) = kv;
            union { uint4 u; _Float16 h[8]; } vv;
            vv.u = *(const uint4*)(Vg + r * DHEAD + c8 * 8);
#pragma unroll
            for (int e = 0; e < 8; e++)
                Vts[(c8 * 8 + e) * 72 + r] = vv.h[e];   // transpose into [dh][key]
        }
        __syncthreads();

        // S = Q * K^T  (16 rows x 64 keys per wave), chunk-outer to keep one
        // B fragment live at a time
        v8f s[4] = {};
#pragma unroll
        for (int c = 0; c < 2; c++) {
#pragma unroll
            for (int t = 0; t < 4; t++) {
                Frag bf;                       // B elem (k=dh, N=key): K[key][dh]
                const int kr = t * 16 + l16;   // key row
                const int base = c * 32 + hi * 8;
                bf.u[0] = *(const uint4*)(&Ks[kr * 72 + base]);
                bf.u[1] = *(const uint4*)(&Ks[kr * 72 + base + 16]);
                s[t] = __builtin_amdgcn_wmma_f32_16x16x32_f16(
                    false, qa[c].v, false, bf.v, (short)0, s[t], false, false);
            }
        }

        // online softmax, fused per-row pass: max -> rescale -> exp in place
#pragma unroll
        for (int j = 0; j < 8; j++) {
            float mx = mi[j];
#pragma unroll
            for (int t = 0; t < 4; t++) mx = fmaxf(mx, s[t][j]);
#pragma unroll
            for (int msk = 1; msk < 16; msk <<= 1)
                mx = fmaxf(mx, __shfl_xor(mx, msk, 32));
            const float corr = __expf(mi[j] - mx);
            mi[j] = mx;
#pragma unroll
            for (int t = 0; t < 5; t++) acc[t][j] *= corr;
#pragma unroll
            for (int t = 0; t < 4; t++) s[t][j] = __expf(s[t][j] - mx);
        }

        // repack P (f32 C-layout) -> f16 A-layout via per-wave LDS region
        _Float16* Pw = &Ps[wave * 16 * 72];
#pragma unroll
        for (int t = 0; t < 4; t++)
#pragma unroll
            for (int j = 0; j < 8; j++)
                Pw[(hi * 8 + j) * 72 + t * 16 + l16] = (_Float16)s[t][j];
        __syncthreads();

        // O += P * V ; chunk-outer so only one P fragment is live
#pragma unroll
        for (int c = 0; c < 2; c++) {
            Frag pa;
            {
                const int base = c * 32 + hi * 8;
                pa.u[0] = *(const uint4*)(&Pw[l16 * 72 + base]);
                pa.u[1] = *(const uint4*)(&Pw[l16 * 72 + base + 16]);
            }
#pragma unroll
            for (int t = 0; t < 5; t++) {      // t==4: ones tile -> row sums
                Frag bf;
                const int vr = t * 16 + l16;   // dh row of Vts (64..79 = ones)
                const int base = c * 32 + hi * 8;
                bf.u[0] = *(const uint4*)(&Vts[vr * 72 + base]);
                bf.u[1] = *(const uint4*)(&Vts[vr * 72 + base + 16]);
                acc[t] = __builtin_amdgcn_wmma_f32_16x16x32_f16(
                    false, pa.v, false, bf.v, (short)0, acc[t], false, false);
            }
        }
    }

    // epilogue: normalize by row sums (replicated in every lane of acc[4])
    const int b = bh >> 3, h = bh & 7;
#pragma unroll
    for (int j = 0; j < 8; j++) {
        const float rinv = 1.f / acc[4][j];
        const int n = qb * 64 + wave * 16 + hi * 8 + j;
#pragma unroll
        for (int t = 0; t < 4; t++) {
            const int o = h * DHEAD + t * 16 + l16;
            O[((size_t)b * NTOK + n) * INNER + o] = (__half)(acc[t][j] * rinv);
        }
    }
}

// ---------------------------------------------------------------------------
// Launch
// ---------------------------------------------------------------------------
extern "C" void kernel_launch(void* const* d_in, const int* in_sizes, int n_in,
                              void* d_out, int out_size, void* d_ws, size_t ws_size,
                              hipStream_t stream)
{
    (void)in_sizes; (void)n_in; (void)out_size; (void)ws_size;

    const float* x = (const float*)d_in[0];
    const float* out_w = (const float*)d_in[25];
    const float* out_b = (const float*)d_in[26];

    char* ws = (char*)d_ws;
    const size_t Y_BYTES   = (size_t)TROWS * DIMC  * 2;  //  6,422,528
    const size_t QKV_BYTES = (size_t)TROWS * INNER * 2;  // 12,845,056
    __half* y   = (__half*)ws;
    __half* qf  = (__half*)(ws + Y_BYTES);
    __half* kf  = (__half*)(ws + Y_BYTES + QKV_BYTES);
    __half* vf  = (__half*)(ws + Y_BYTES + 2 * QKV_BYTES);
    __half* ao  = (__half*)(ws + Y_BYTES + 3 * QKV_BYTES);

    __half* qkv_out[3] = { qf, kf, vf };
    const float alphas[3] = { 0.125f /* DH^-0.5 folded into q */, 1.f, 1.f };

    for (int br = 0; br < 3; br++) {
        const int base = 1 + br * 8;
        dwconv_bn_gelu<<<TROWS, 256, 0, stream>>>(
            x,
            (const float*)d_in[base + 0],  // dw_w
            (const float*)d_in[base + 1],  // dw_b
            (const float*)d_in[base + 2],  // bn_g
            (const float*)d_in[base + 3],  // bn_b
            (const float*)d_in[base + 4],  // bn_m
            (const float*)d_in[base + 5],  // bn_v
            y);
        gemm_wmma<DIMC, INNER, 0><<<dim3(TROWS / 64, INNER / 64), 128, 0, stream>>>(
            y,
            (const float*)d_in[base + 6],  // pw_w [512,256]
            (const float*)d_in[base + 7],  // pw_b
            (void*)qkv_out[br], alphas[br]);
    }

    flash_attn<<<BATCH * NHEAD * KBLKS, 128, 0, stream>>>(qf, kf, vf, ao);

    gemm_wmma<INNER, DIMC, 1><<<dim3(TROWS / 64, DIMC / 64), 128, 0, stream>>>(
        ao, out_w, out_b, d_out, 1.f);
}
